// SlotAttentionWithFeedbackHyper_58076547776906
// MI455X (gfx1250) — matrix operand order repaired and checked by reading, not compiled, not run on hardware
//
#include <hip/hip_runtime.h>
#include <hip/hip_bf16.h>

typedef __attribute__((ext_vector_type(16))) _Float16 v16h;
typedef __attribute__((ext_vector_type(8)))  _Float16 v8h;
typedef __attribute__((ext_vector_type(8)))  float    v8f;

#define Bsz 32
#define Nn  4096
#define Ss  8
#define Dd  256
#define Hh  4
#define DHd 64
#define DOWNd 64
#define EPSv 1e-8f
#define LNEPS 1e-5f

// CDNA5 async global->LDS copy (16B per lane), tracked by ASYNCcnt.
// vdst = LDS byte offset VGPR, vaddr = 64-bit global address, saddr = off.
#define ASYNC_B128(ldsoff, gptr)                                              \
    asm volatile("global_load_async_to_lds_b128 %0, %1, off"                  \
                 :: "v"(ldsoff), "v"(gptr) : "memory")

static __device__ __forceinline__ void wait_async0() {
#if __has_builtin(__builtin_amdgcn_s_wait_asynccnt)
    __builtin_amdgcn_s_wait_asynccnt(0);
#else
    asm volatile("s_wait_asynccnt 0x0" ::: "memory");
#endif
}

// ---------------------------------------------------------------------------
// Kernel 1: LayerNorm(inputs) -> k_inputs (f32 out), A_f16 (ws), v_inputs (f32 out)
// grid = B*N blocks, 256 threads (one per d)
// ---------------------------------------------------------------------------
__global__ __launch_bounds__(256) void k_ln(const float* __restrict__ x,
    const float* __restrict__ nw, const float* __restrict__ nb,
    const float* __restrict__ lnw, const float* __restrict__ lnb,
    float* __restrict__ k_out, float* __restrict__ v_out,
    _Float16* __restrict__ a16)
{
    int row = blockIdx.x;              // b*N + n
    int b = row >> 12;
    int n = row & (Nn - 1);
    int t = threadIdx.x;
    float xv = x[(size_t)row * Dd + t];

    __shared__ float red[256];
    __shared__ float smean, sinv;

    red[t] = xv; __syncthreads();
    for (int s = 128; s > 0; s >>= 1) { if (t < s) red[t] += red[t + s]; __syncthreads(); }
    if (t == 0) smean = red[0] * (1.0f / 256.0f);
    __syncthreads();
    float dv = xv - smean;
    red[t] = dv * dv; __syncthreads();
    for (int s = 128; s > 0; s >>= 1) { if (t < s) red[t] += red[t + s]; __syncthreads(); }
    if (t == 0) sinv = rsqrtf(red[0] * (1.0f / 256.0f) + LNEPS);
    __syncthreads();

    float ln = dv * sinv;
    float kin = ln * nw[t] + nb[t];
    k_out[(size_t)row * Dd + t] = kin;
    a16[(size_t)row * Dd + t] = (_Float16)kin;

    #pragma unroll
    for (int ks = 0; ks < Ss; ++ks) {
        size_t wi = ((size_t)b * Ss + ks) * Dd + t;
        size_t vo = (((size_t)b * Ss + ks) * Nn + n) * Dd + t;
        v_out[vo] = ln * lnw[wi] + lnb[wi];
    }
}

// ---------------------------------------------------------------------------
// Kernel 2: transpose Wk/Wv (f32 [O][D]) -> f16 [D][O] for WMMA B fragments
// ---------------------------------------------------------------------------
__global__ __launch_bounds__(256) void k_wt(const float* __restrict__ Wk,
    const float* __restrict__ Wv, _Float16* __restrict__ WkT, _Float16* __restrict__ WvT)
{
    int which = blockIdx.x >> 8;
    int d = blockIdx.x & 255;
    int o = threadIdx.x;
    const float* W = which ? Wv : Wk;
    _Float16* T = which ? WvT : WkT;
    T[(size_t)d * Dd + o] = (_Float16)W[(size_t)o * Dd + d];
}

// ---------------------------------------------------------------------------
// Kernel 3: C[M,256] = A[M,256] x Bt[256,256]   (f16 in, f32 WMMA accum, f16 out)
// block = 256 thr = 8 waves (2 M-rows x 4 N-cols), wave tile 16x64.
// B chunks (32x256 f16 = 16 KB) staged in LDS via async DMA, double buffered;
// DMA of chunk i+1 overlaps the 4 WMMAs of chunk i.  K loop fully unrolled so
// accumulators stay pinned (no v_mov shuffles / hazard NOPs between WMMAs).
// ---------------------------------------------------------------------------
__global__ __launch_bounds__(256) void gemm_f16(const _Float16* __restrict__ A,
    const _Float16* __restrict__ Bt, _Float16* __restrict__ C)
{
    __shared__ __align__(64) _Float16 bbuf[2][32 * 256];   // 2 x 16 KB

    const int K = 256;
    int t = threadIdx.x;
    int wave = t >> 5;
    int lane = t & 31;
    int wr = wave >> 2;            // 0..1
    int wc = wave & 3;             // 0..3
    int mbase = blockIdx.x * 32 + wr * 16;
    int nbase = wc * 64;
    int mrow  = mbase + (lane & 15);
    int khalf = (lane >> 4) * 8;   // 0 or 8

    // async-copy one 32x256 chunk of Bt into bbuf[buf]; 4 x 16B per thread
    uint32_t lbase0 = (uint32_t)(size_t)(&bbuf[0][0]);
    uint32_t lbase1 = (uint32_t)(size_t)(&bbuf[1][0]);
    auto issue_chunk = [&](int k0, int buf) {
        const _Float16* g = Bt + (size_t)k0 * Dd;
        uint32_t lb = buf ? lbase1 : lbase0;
        #pragma unroll
        for (int u = 0; u < 4; ++u) {
            int unit = t + u * 256;                 // 16-byte unit index
            ASYNC_B128(lb + (uint32_t)unit * 16, g + (size_t)unit * 8);
        }
    };

    issue_chunk(0, 0);   // prologue

    v8f acc0 = {}, acc1 = {}, acc2 = {}, acc3 = {};

    #pragma unroll
    for (int i = 0; i < 8; ++i) {
        wait_async0();        // own writes to bbuf[i&1] have landed
        __syncthreads();      // everyone's writes landed; prior reads consumed
        if (i < 7) issue_chunk((i + 1) * 32, (i + 1) & 1);   // overlap DMA

        int k0 = i * 32;
        // A fragment: lane m holds K = {k0+khalf..+7} U {k0+16+khalf..+7}
        const _Float16* ap = A + (size_t)mrow * K + k0 + khalf;
        v8h a_lo = *(const v8h*)ap;
        v8h a_hi = *(const v8h*)(ap + 16);
        v16h a;
        #pragma unroll
        for (int j = 0; j < 8; ++j) { a[j] = a_lo[j]; a[8 + j] = a_hi[j]; }
        if (i < 7)
            __builtin_prefetch(ap + 32, 0, 0);   // global_prefetch_b8 next A chunk

        // B fragments from LDS: lane l holds column n=l&15, K = {khalf..+7} U {16+khalf..+7}
        const _Float16* bp = &bbuf[i & 1][khalf * Dd + nbase + (lane & 15)];
        v16h b0, b1, b2, b3;
        #pragma unroll
        for (int j = 0; j < 8; ++j) {
            b0[j] = bp[j * Dd];      b0[8 + j] = bp[(16 + j) * Dd];
            b1[j] = bp[j * Dd + 16]; b1[8 + j] = bp[(16 + j) * Dd + 16];
            b2[j] = bp[j * Dd + 32]; b2[8 + j] = bp[(16 + j) * Dd + 32];
            b3[j] = bp[j * Dd + 48]; b3[8 + j] = bp[(16 + j) * Dd + 48];
        }

        acc0 = __builtin_amdgcn_wmma_f32_16x16x32_f16(false, a, false, b0, (short)0, acc0, false, false);
        acc1 = __builtin_amdgcn_wmma_f32_16x16x32_f16(false, a, false, b1, (short)0, acc1, false, false);
        acc2 = __builtin_amdgcn_wmma_f32_16x16x32_f16(false, a, false, b2, (short)0, acc2, false, false);
        acc3 = __builtin_amdgcn_wmma_f32_16x16x32_f16(false, a, false, b3, (short)0, acc3, false, false);
    }

    // C layout: VGPR r, lane l -> M = r + (l>>4)*8, N = l&15
    int mo = mbase + (lane >> 4) * 8;
    int no = nbase + (lane & 15);
    #pragma unroll
    for (int r = 0; r < 8; ++r) {
        size_t o = (size_t)(mo + r) * Dd + no;
        C[o]      = (_Float16)acc0[r];
        C[o + 16] = (_Float16)acc1[r];
        C[o + 32] = (_Float16)acc2[r];
        C[o + 48] = (_Float16)acc3[r];
    }
}

// ---------------------------------------------------------------------------
// Kernel 4: slots LN + q projection.  grid = B*S blocks, 256 threads (o)
// ---------------------------------------------------------------------------
__global__ __launch_bounds__(256) void k_q(const float* __restrict__ slots,
    const float* __restrict__ nsw, const float* __restrict__ nsb,
    const float* __restrict__ Wq, float* __restrict__ q)
{
    int row = blockIdx.x;
    int t = threadIdx.x;
    __shared__ float red[256];
    __shared__ float sn[256];
    __shared__ float smean, sinv;

    float xv = slots[(size_t)row * Dd + t];
    red[t] = xv; __syncthreads();
    for (int s = 128; s > 0; s >>= 1) { if (t < s) red[t] += red[t + s]; __syncthreads(); }
    if (t == 0) smean = red[0] * (1.0f / 256.0f);
    __syncthreads();
    float dv = xv - smean;
    red[t] = dv * dv; __syncthreads();
    for (int s = 128; s > 0; s >>= 1) { if (t < s) red[t] += red[t + s]; __syncthreads(); }
    if (t == 0) sinv = rsqrtf(red[0] * (1.0f / 256.0f) + LNEPS);
    __syncthreads();
    sn[t] = dv * sinv * nsw[t] + nsb[t];
    __syncthreads();

    float acc = 0.0f;
    const float* wrow = Wq + (size_t)t * Dd;
    for (int d = 0; d < Dd; ++d) acc += sn[d] * wrow[d];
    q[(size_t)row * Dd + t] = acc;
}

// ---------------------------------------------------------------------------
// Kernel 5: dots + softmax over (s,h)=32 + attn_out + row-sum atomics
// grid = (B, N/128), 256 threads (2 threads per n: each 2 heads x 8 slots)
// ---------------------------------------------------------------------------
__global__ __launch_bounds__(256) void k_dots(const float* __restrict__ q,
    const _Float16* __restrict__ kf, float* __restrict__ attn,
    float* __restrict__ attn_out, float* __restrict__ sums)
{
    int b = blockIdx.x;
    int n0 = blockIdx.y * 128;
    int t = threadIdx.x;

    __shared__ float qs[2048];        // q[b] : [s*4+h][64]
    __shared__ float sd[128][32];
    __shared__ float ssum[32];

    for (int i = t; i < 2048; i += 256) qs[i] = q[(size_t)b * 2048 + i];
    if (t < 32) ssum[t] = 0.0f;
    __syncthreads();

    int nl = t >> 1;
    int half = t & 1;
    int n = n0 + nl;
    const _Float16* krow = kf + ((size_t)b * Nn + n) * Dd;

    #pragma unroll
    for (int hh = 0; hh < 2; ++hh) {
        int h = half * 2 + hh;
        float accs[8] = {0,0,0,0,0,0,0,0};
        for (int d = 0; d < DHd; ++d) {
            float kv = (float)krow[h * DHd + d];
            #pragma unroll
            for (int s = 0; s < 8; ++s) accs[s] += qs[(s * 4 + h) * DHd + d] * kv;
        }
        #pragma unroll
        for (int s = 0; s < 8; ++s) sd[nl][s * 4 + h] = accs[s] * 0.125f; // 64^-0.5
    }
    __syncthreads();

    float av[32];
    if (t < 128) {
        float mx = -1e30f;
        #pragma unroll
        for (int i = 0; i < 32; ++i) { av[i] = sd[t][i]; mx = fmaxf(mx, av[i]); }
        float sm = 0.0f;
        #pragma unroll
        for (int i = 0; i < 32; ++i) { av[i] = expf(av[i] - mx); sm += av[i]; }
        float inv = 1.0f / sm;
        #pragma unroll
        for (int i = 0; i < 32; ++i) av[i] *= inv;

        int n2 = n0 + t;
        for (int sh = 0; sh < 32; ++sh)
            attn[((size_t)b * 32 + sh) * Nn + n2] = av[sh];
        #pragma unroll
        for (int s = 0; s < 8; ++s) {
            float m4 = (av[s*4] + av[s*4+1] + av[s*4+2] + av[s*4+3]) * 0.25f;
            attn_out[((size_t)b * Ss + s) * Nn + n2] = m4;
        }
        for (int sh = 0; sh < 32; ++sh) atomicAdd(&ssum[sh], av[sh]);
    }
    __syncthreads();
    if (t < 32) atomicAdd(&sums[b * 32 + t], ssum[t]);
}

// ---------------------------------------------------------------------------
// Kernel 6: updates[b,s,h,dh] = sum_n (attn+EPS)*v / (sum+N*EPS)
// grid = B*32 blocks (b,sh), 256 thr = 4 n-groups x 64 dh
// ---------------------------------------------------------------------------
__global__ __launch_bounds__(256) void k_upd(const _Float16* __restrict__ vf,
    const float* __restrict__ attn, const float* __restrict__ sums,
    float* __restrict__ updates)
{
    int blk = blockIdx.x;
    int b = blk >> 5;
    int sh = blk & 31;
    int s = sh >> 2, h = sh & 3;
    int t = threadIdx.x;
    int g = t >> 6;
    int dh = t & 63;

    const float* arow = attn + ((size_t)b * 32 + sh) * Nn;
    const _Float16* vbase = vf + (size_t)b * Nn * Dd + h * DHd + dh;

    __shared__ float atile[256];
    __shared__ float red[256];

    float acc = 0.0f;
    for (int n0 = 0; n0 < Nn; n0 += 256) {
        atile[t] = arow[n0 + t];
        __syncthreads();
        for (int nn = g; nn < 256; nn += 4)
            acc += (atile[nn] + EPSv) * (float)vbase[(size_t)(n0 + nn) * Dd];
        __syncthreads();
    }
    red[t] = acc;
    __syncthreads();
    if (t < 64) {
        float tot = red[t] + red[t + 64] + red[t + 128] + red[t + 192];
        float den = sums[b * 32 + sh] + (float)Nn * EPSv;
        updates[((size_t)b * Ss + s) * Dd + h * DHd + t] = tot / den;
    }
}

// ---------------------------------------------------------------------------
// Kernel 7: GRU cell.  grid = B*S blocks (rows), 256 threads (gate index)
// ---------------------------------------------------------------------------
__global__ __launch_bounds__(256) void k_gru(const float* __restrict__ upd,
    const float* __restrict__ hprev, const float* __restrict__ wih,
    const float* __restrict__ whh, const float* __restrict__ bih,
    const float* __restrict__ bhh, float* __restrict__ hnew)
{
    int m = blockIdx.x;
    int t = threadIdx.x;
    __shared__ float xs[256], hs[256];
    xs[t] = upd[(size_t)m * Dd + t];
    hs[t] = hprev[(size_t)m * Dd + t];
    __syncthreads();

    float xr = bih[t], xz = bih[256 + t], xn = bih[512 + t];
    float hr = bhh[t], hz = bhh[256 + t], hn = bhh[512 + t];
    const float* wr_ = wih + (size_t)t * Dd;
    const float* wz_ = wih + (size_t)(256 + t) * Dd;
    const float* wn_ = wih + (size_t)(512 + t) * Dd;
    const float* ur_ = whh + (size_t)t * Dd;
    const float* uz_ = whh + (size_t)(256 + t) * Dd;
    const float* un_ = whh + (size_t)(512 + t) * Dd;
    for (int d = 0; d < Dd; ++d) {
        float xd = xs[d], hd = hs[d];
        xr += xd * wr_[d]; xz += xd * wz_[d]; xn += xd * wn_[d];
        hr += hd * ur_[d]; hz += hd * uz_[d]; hn += hd * un_[d];
    }
    float r = 1.0f / (1.0f + expf(-(xr + hr)));
    float z = 1.0f / (1.0f + expf(-(xz + hz)));
    float nn = tanhf(xn + r * hn);
    hnew[(size_t)m * Dd + t] = (1.0f - z) * nn + z * hs[t];
}

// ---------------------------------------------------------------------------
// Kernel 8: conditional adapter: LN + down-proj + exact GELU + up-proj + residual
// ---------------------------------------------------------------------------
__global__ __launch_bounds__(256) void k_adapter(const float* __restrict__ slots_in,
    const float* __restrict__ aw, const float* __restrict__ ab,
    const float* __restrict__ dproj, const float* __restrict__ uproj,
    float* __restrict__ slots_out, float* __restrict__ out_final)
{
    int row = blockIdx.x;           // b*8 + s
    int b = row >> 3;
    int t = threadIdx.x;
    __shared__ float red[256];
    __shared__ float nrm[256];
    __shared__ float gel[64];
    __shared__ float smean, sinv;

    float xv = slots_in[(size_t)row * Dd + t];
    red[t] = xv; __syncthreads();
    for (int s = 128; s > 0; s >>= 1) { if (t < s) red[t] += red[t + s]; __syncthreads(); }
    if (t == 0) smean = red[0] * (1.0f / 256.0f);
    __syncthreads();
    float dv = xv - smean;
    red[t] = dv * dv; __syncthreads();
    for (int s = 128; s > 0; s >>= 1) { if (t < s) red[t] += red[t + s]; __syncthreads(); }
    if (t == 0) sinv = rsqrtf(red[0] * (1.0f / 256.0f) + LNEPS);
    __syncthreads();
    nrm[t] = dv * sinv * aw[(size_t)b * Dd + t] + ab[(size_t)b * Dd + t];
    __syncthreads();

    if (t < DOWNd) {
        float accd = 0.0f;
        const float* dp = dproj + (size_t)b * Dd * DOWNd + t;   // [b][d][e]
        for (int d = 0; d < Dd; ++d) accd += nrm[d] * dp[(size_t)d * DOWNd];
        gel[t] = 0.5f * accd * (1.0f + erff(accd * 0.70710678118654752f));
    }
    __syncthreads();

    float accu = 0.0f;
    const float* up = uproj + (size_t)b * DOWNd * Dd + t;       // [b][e][d]
    for (int e = 0; e < DOWNd; ++e) accu += gel[e] * up[(size_t)e * Dd];
    float res = accu + xv;
    slots_out[(size_t)row * Dd + t] = res;
    out_final[(size_t)row * Dd + t] = res;
}

// ---------------------------------------------------------------------------
extern "C" void kernel_launch(void* const* d_in, const int* in_sizes, int n_in,
                              void* d_out, int out_size, void* d_ws, size_t ws_size,
                              hipStream_t stream)
{
    const float* inputs  = (const float*)d_in[0];
    const float* cond    = (const float*)d_in[1];
    const float* ln_w    = (const float*)d_in[2];
    const float* ln_b    = (const float*)d_in[3];
    const float* ada_up  = (const float*)d_in[4];
    const float* ada_dn  = (const float*)d_in[5];
    const float* ada_lw  = (const float*)d_in[6];
    const float* ada_lb  = (const float*)d_in[7];
    const float* Wq      = (const float*)d_in[8];
    const float* Wk      = (const float*)d_in[9];
    const float* Wv      = (const float*)d_in[10];
    const float* nin_w   = (const float*)d_in[11];
    const float* nin_b   = (const float*)d_in[12];
    const float* nsl_w   = (const float*)d_in[13];
    const float* nsl_b   = (const float*)d_in[14];
    const float* gru_wih = (const float*)d_in[15];
    const float* gru_whh = (const float*)d_in[16];
    const float* gru_bih = (const float*)d_in[17];
    const float* gru_bhh = (const float*)d_in[18];

    float* out = (float*)d_out;
    float* o_slots = out;                                    // [B,S,D]
    float* o_attn  = out + (size_t)Bsz * Ss * Dd;            // [B,S,N]
    float* o_kin   = o_attn + (size_t)Bsz * Ss * Nn;         // [B,N,D]
    float* o_vin   = o_kin + (size_t)Bsz * Nn * Dd;          // [B,S,N,D]

    char* ws = (char*)d_ws;
    size_t off = 0;
    auto alloc = [&](size_t bytes) -> void* {
        void* p = ws + off;
        off = (off + bytes + 255) & ~(size_t)255;
        return p;
    };
    _Float16* A16   = (_Float16*)alloc((size_t)Bsz * Nn * Dd * 2);
    _Float16* WkT   = (_Float16*)alloc((size_t)Dd * Dd * 2);
    _Float16* WvT   = (_Float16*)alloc((size_t)Dd * Dd * 2);
    _Float16* Kf16  = (_Float16*)alloc((size_t)Bsz * Nn * Dd * 2);
    _Float16* Vf16  = (_Float16*)alloc((size_t)Bsz * Nn * Dd * 2);
    float* attn_ws  = (float*)alloc((size_t)Bsz * 32 * Nn * 4);
    float* sums     = (float*)alloc((size_t)Bsz * 32 * 4);
    float* slots0   = (float*)alloc((size_t)Bsz * Ss * Dd * 4);
    float* slots1   = (float*)alloc((size_t)Bsz * Ss * Dd * 4);
    float* qbuf     = (float*)alloc((size_t)Bsz * Ss * Dd * 4);
    float* updbuf   = (float*)alloc((size_t)Bsz * Ss * Dd * 4);

    // slots <- conditioning
    hipMemcpyAsync(slots0, cond, (size_t)Bsz * Ss * Dd * 4,
                   hipMemcpyDeviceToDevice, stream);

    // LN pass: k_inputs, v_inputs, f16 GEMM operand
    k_ln<<<Bsz * Nn, 256, 0, stream>>>(inputs, nin_w, nin_b, ln_w, ln_b,
                                       o_kin, o_vin, A16);
    // weight transposes to f16
    k_wt<<<512, 256, 0, stream>>>(Wk, Wv, WkT, WvT);

    // k / v projections via WMMA with async-LDS double-buffered B panel
    gemm_f16<<<(Bsz * Nn) / 32, 256, 0, stream>>>(A16, WkT, Kf16);
    gemm_f16<<<(Bsz * Nn) / 32, 256, 0, stream>>>(A16, WvT, Vf16);

    for (int it = 0; it < 3; ++it) {
        hipMemsetAsync(sums, 0, (size_t)Bsz * 32 * 4, stream);
        k_q<<<Bsz * Ss, 256, 0, stream>>>(slots0, nsl_w, nsl_b, Wq, qbuf);
        k_dots<<<dim3(Bsz, Nn / 128), 256, 0, stream>>>(qbuf, Kf16, attn_ws,
                                                        o_attn, sums);
        k_upd<<<Bsz * 32, 256, 0, stream>>>(Vf16, attn_ws, sums, updbuf);
        k_gru<<<Bsz * Ss, 256, 0, stream>>>(updbuf, slots0, gru_wih, gru_whh,
                                            gru_bih, gru_bhh, slots1);
        k_adapter<<<Bsz * Ss, 256, 0, stream>>>(slots1, ada_lw, ada_lb,
                                                ada_dn, ada_up, slots0, o_slots);
    }
}
